// GraphTransformerEncoderLayer_32770600468932
// MI455X (gfx1250) — matrix operand order repaired and checked by reading, not compiled, not run on hardware
//
#include <hip/hip_runtime.h>
#include <hip/hip_bf16.h>
#include <math.h>

// ---------------------------------------------------------------------------
// GraphTransformerEncoderLayer for MI455X (gfx1250), bf16 WMMA pipeline.
// B=8 S=512 D=512 H=8 R=4 FF=2048 HD=64
// Round 3: TDM-stage the attention K tiles (double-buffered), dword-combine
// the relation-mask loads, vectorize V^T staging loads.
// ---------------------------------------------------------------------------

typedef __bf16 bf16_t;
typedef __attribute__((ext_vector_type(16))) __bf16 v16bf;
typedef __attribute__((ext_vector_type(8)))  __bf16 v8bf;
typedef __attribute__((ext_vector_type(8)))  float  v8f;

union F8 { v8f v; float f[8]; };
union AF { v16bf v; v8bf h[2]; };
union V4B { uint64_t u; bf16_t b[4]; };

constexpr int BATCH = 8, SEQ = 512, DM = 512, NH = 8, NREL = 4, FFD = 2048, HDIM = 64;
constexpr int MROWS = BATCH * SEQ;                 // 4096
constexpr float ATT_SCALE = 0.125f;                // 1/sqrt(64)
constexpr float LN_EPS = 1e-5f;

// ---------------- TDM support detection --------------------------------
#if defined(__has_builtin)
#if __has_builtin(__builtin_amdgcn_tensor_load_to_lds)
#define HAVE_TDM 1
#endif
#endif
#if defined(__has_include)
#if __has_include(<hip/amd_detail/amd_gfx1250_TDM.h>)
#define TDM_6ARG 1   // therock-10.0 headers => clang-23 6-arg builtin
#endif
#endif

#if defined(HAVE_TDM)
typedef uint32_t u32x4 __attribute__((ext_vector_type(4)));
typedef int      i32x4 __attribute__((ext_vector_type(4)));
typedef int      i32x8 __attribute__((ext_vector_type(8)));

// Issue a TDM 2D tile load: global (row-major, ld_elems bf16 per row) -> LDS
// packed tile [tile_rows][tile_k] bf16. Descriptor per CDNA5 ISA 8.3/8.4.
__device__ __forceinline__ void tdm_load_2d(uint32_t lds_off, const void* gaddr,
                                            uint32_t tile_k, uint32_t tile_rows,
                                            uint32_t ld_elems, uint32_t nrows) {
  const uint64_t ga = (uint64_t)(uintptr_t)gaddr;
  u32x4 g0;
  g0[0] = 1u;                                            // count=1, user mode
  g0[1] = lds_off;                                       // lds_addr (bytes)
  g0[2] = (uint32_t)ga;                                  // global_addr[31:0]
  g0[3] = (uint32_t)((ga >> 32) & 0x01FFFFFFu) | (2u << 30);  // ga[56:32]|type=2
  i32x8 g1;
  g1[0] = (int)(1u << 16);                               // data_size=1 (2 bytes)
  g1[1] = (int)((ld_elems & 0xFFFFu) << 16);             // tensor_dim0 lo
  g1[2] = (int)((ld_elems >> 16) | ((nrows & 0xFFFFu) << 16));  // dim0 hi|dim1 lo
  g1[3] = (int)((nrows >> 16) | (tile_k << 16));         // dim1 hi | tile_dim0
  g1[4] = (int)(tile_rows & 0xFFFFu);                    // tile_dim1 (tile_dim2=0)
  g1[5] = (int)ld_elems;                                 // tensor_dim0_stride lo
  g1[6] = 0;                                             // stride hi | dim1_stride
  g1[7] = 0;
  const i32x4 gz = {0, 0, 0, 0};
#if defined(TDM_6ARG)
  const i32x8 gz8 = {0, 0, 0, 0, 0, 0, 0, 0};
  __builtin_amdgcn_tensor_load_to_lds(g0, g1, gz, gz, gz8, 0);
#else
  __builtin_amdgcn_tensor_load_to_lds(g0, g1, gz, gz, 0);
#endif
}

__device__ __forceinline__ void wait_tensorcnt0() {
#if __has_builtin(__builtin_amdgcn_s_wait_tensorcnt)
  __builtin_amdgcn_s_wait_tensorcnt(0);
#else
  asm volatile("s_wait_tensorcnt 0x0" ::: "memory");
#endif
}
#endif  // HAVE_TDM

// ---- WMMA helper: D = A(16x32 bf16) * B(32x16 bf16) + C(16x16 f32) ----
__device__ __forceinline__ v8f wmma_bf16(v16bf a, v16bf b, v8f c) {
  return __builtin_amdgcn_wmma_f32_16x16x32_bf16(false, a, false, b, (short)0, c,
                                                 false, false);
}

// A fragment from row-major [M][K] tile origin p (16x32 tile, leading dim ld).
__device__ __forceinline__ v16bf frag_a(const bf16_t* p, int ld) {
  const int lane = threadIdx.x & 31;
  const int row = lane & 15;
  const int kb  = (lane >> 4) << 3;          // 0 or 8
  AF u;
  u.h[0] = *(const v8bf*)(p + row * ld + kb);
  u.h[1] = *(const v8bf*)(p + row * ld + kb + 16);
  return u.v;
}

// B fragment (32x16, K x N) from a row-major [N][K] matrix tile origin p.
__device__ __forceinline__ v16bf frag_b(const bf16_t* p, int ld) {
  const int lane = threadIdx.x & 31;
  const int n  = lane & 15;
  const int kb = (lane >> 4) << 3;           // 0 or 8
  AF u;
  u.h[0] = *(const v8bf*)(p + n * ld + kb);
  u.h[1] = *(const v8bf*)(p + n * ld + kb + 16);
  return u.v;
}

// ---------------------------------------------------------------------------
// f32 -> bf16 weight conversion
// ---------------------------------------------------------------------------
__global__ __launch_bounds__(256) void cvt_bf16(const float* __restrict__ in,
                                                bf16_t* __restrict__ out, int n) {
  int i = blockIdx.x * 256 + threadIdx.x;
  if (i < n) out[i] = (bf16_t)in[i];
}

// rel_mat [r][h][d][e] f32 -> relT [r][h][e][d] bf16 (so it loads as B operand)
__global__ __launch_bounds__(256) void cvt_rel(const float* __restrict__ rel,
                                               bf16_t* __restrict__ out) {
  int i = blockIdx.x * 256 + threadIdx.x;          // NREL*NH*64*64 = 131072
  int rh = i >> 12;
  int d = (i >> 6) & 63;
  int e = i & 63;
  out[(size_t)rh * 4096 + e * 64 + d] = (bf16_t)rel[(size_t)rh * 4096 + d * 64 + e];
}

// ---------------------------------------------------------------------------
// LayerNorm (f32 in) -> bf16 out. One wave32 per row of 512.
// ---------------------------------------------------------------------------
__global__ __launch_bounds__(256) void ln_bf16(const float* __restrict__ x,
                                               const float* __restrict__ w,
                                               const float* __restrict__ bvec,
                                               bf16_t* __restrict__ out) {
  const int wid = threadIdx.x >> 5, lane = threadIdx.x & 31;
  const int row = blockIdx.x * 8 + wid;
  const float* xr = x + (size_t)row * DM;
  float vals[16];
  float s = 0.f;
#pragma unroll
  for (int c = 0; c < 16; ++c) { vals[c] = xr[c * 32 + lane]; s += vals[c]; }
#pragma unroll
  for (int o = 16; o >= 1; o >>= 1) s += __shfl_xor(s, o, 32);
  const float mean = s * (1.0f / DM);
  float vs = 0.f;
#pragma unroll
  for (int c = 0; c < 16; ++c) { float d = vals[c] - mean; vs += d * d; }
#pragma unroll
  for (int o = 16; o >= 1; o >>= 1) vs += __shfl_xor(vs, o, 32);
  const float rstd = rsqrtf(vs * (1.0f / DM) + LN_EPS);
#pragma unroll
  for (int c = 0; c < 16; ++c) {
    int col = c * 32 + lane;
    out[(size_t)row * DM + col] = (bf16_t)((vals[c] - mean) * rstd * w[col] + bvec[col]);
  }
}

// ---------------------------------------------------------------------------
// Generic WMMA GEMM:  C[M,N] = A[M,K](bf16) @ Bw[N,K]^T(bf16) + bias, epilogue:
//  EPI 0: store bf16            (QKV)
//  EPI 1: +resid, valid-mask, store f32   (attn out-proj)
//  EPI 2: exact GELU, store bf16          (FFN up)
//  EPI 3: +resid, valid-mask, store f32   (FFN down -> d_out)
// Block tile 128x128, 8 wave32 waves (4x2), wave tile 32x64, K-step 32.
// With TDM: wave 0 DMA-stages double-buffered 128x32 A/B tiles into LDS
// (TENSORcnt + workgroup barrier hand-off); compute waves read LDS only.
// ---------------------------------------------------------------------------
template <int EPI>
__global__ __launch_bounds__(256) void gemm_bf16_wmma(
    const bf16_t* __restrict__ A, const bf16_t* __restrict__ Bw,
    const float* __restrict__ bias, void* __restrict__ Cout,
    const float* __restrict__ resid, const unsigned char* __restrict__ vmask,
    int N, int K) {
  const int wid = threadIdx.x >> 5;
  const int lane = threadIdx.x & 31;
  const int wm = wid >> 1, wn = wid & 1;
  const int row0 = blockIdx.y * 128 + wm * 32;
  const int col0 = blockIdx.x * 128 + wn * 64;

  F8 acc[2][4] = {};

#if defined(HAVE_TDM)
  __shared__ __align__(16) bf16_t atile[2][128][32];   // 8 KB x2
  __shared__ __align__(16) bf16_t btile[2][128][32];   // 8 KB x2
  const bf16_t* Ablk = A + (size_t)(blockIdx.y * 128) * K;
  const bf16_t* Bblk = Bw + (size_t)(blockIdx.x * 128) * K;
  if (threadIdx.x < 32) {   // wave 0 is the DMA producer
    tdm_load_2d((uint32_t)(uintptr_t)&atile[0][0][0], Ablk, 32, 128, (uint32_t)K,
                (uint32_t)MROWS);
    tdm_load_2d((uint32_t)(uintptr_t)&btile[0][0][0], Bblk, 32, 128, (uint32_t)K,
                (uint32_t)N);
  }
  for (int kk = 0; kk < K; kk += 32) {
    const int buf = (kk >> 5) & 1;
    if (threadIdx.x < 32) wait_tensorcnt0();     // current buffers resident
    __syncthreads();
    if (threadIdx.x < 32 && kk + 32 < K) {       // prefetch next K-step tiles
      tdm_load_2d((uint32_t)(uintptr_t)&atile[buf ^ 1][0][0], Ablk + kk + 32, 32,
                  128, (uint32_t)K, (uint32_t)MROWS);
      tdm_load_2d((uint32_t)(uintptr_t)&btile[buf ^ 1][0][0], Bblk + kk + 32, 32,
                  128, (uint32_t)K, (uint32_t)N);
    }
    v16bf a0 = frag_a(&atile[buf][wm * 32][0], 32);
    v16bf a1 = frag_a(&atile[buf][wm * 32 + 16][0], 32);
#pragma unroll
    for (int nt = 0; nt < 4; ++nt) {
      v16bf b = frag_b(&btile[buf][wn * 64 + nt * 16][0], 32);
      acc[0][nt].v = wmma_bf16(a0, b, acc[0][nt].v);
      acc[1][nt].v = wmma_bf16(a1, b, acc[1][nt].v);
    }
    __syncthreads();   // all waves done with buf before it is re-filled
  }
#else
  const bf16_t* Arow0 = A + (size_t)row0 * K;
  const bf16_t* Arow1 = A + (size_t)(row0 + 16) * K;
  for (int kk = 0; kk < K; kk += 32) {
    __builtin_prefetch(Arow0 + kk + 256, 0, 0);
    v16bf a0 = frag_a(Arow0 + kk, K);
    v16bf a1 = frag_a(Arow1 + kk, K);
#pragma unroll
    for (int nt = 0; nt < 4; ++nt) {
      v16bf b = frag_b(Bw + (size_t)(col0 + nt * 16) * K + kk, K);
      acc[0][nt].v = wmma_bf16(a0, b, acc[0][nt].v);
      acc[1][nt].v = wmma_bf16(a1, b, acc[1][nt].v);
    }
  }
#endif

  const int cn = lane & 15, hi = lane >> 4;
#pragma unroll
  for (int i = 0; i < 2; ++i)
#pragma unroll
    for (int nt = 0; nt < 4; ++nt)
#pragma unroll
      for (int vv = 0; vv < 8; ++vv) {
        const int row = row0 + i * 16 + vv + hi * 8;
        const int col = col0 + nt * 16 + cn;
        float val = acc[i][nt].f[vv] + bias[col];
        if constexpr (EPI == 0) {
          ((bf16_t*)Cout)[(size_t)row * N + col] = (bf16_t)val;
        } else if constexpr (EPI == 2) {
          val = 0.5f * val * (1.0f + erff(val * 0.70710678118f));
          ((bf16_t*)Cout)[(size_t)row * N + col] = (bf16_t)val;
        } else {
          val += resid[(size_t)row * N + col];
          if (!vmask[row]) val = 0.f;
          ((float*)Cout)[(size_t)row * N + col] = val;
        }
      }
}

// ---------------------------------------------------------------------------
// Fused flash attention with relation-bilinear bias.
// Grid: (SEQ/128, NH, BATCH), 256 threads = 8 waves; each wave owns 16 query
// rows. Per (b,h): V^T staged in LDS; per wave: qm_r = q @ M_r staged in LDS.
// K tiles (32 rows/step) are TDM-staged into a double buffer shared by all 8
// waves (was: 8x redundant global reads). Scores never touch global memory.
// ---------------------------------------------------------------------------
__global__ __launch_bounds__(256) void attn_fused(
    const bf16_t* __restrict__ q, const bf16_t* __restrict__ k,
    const bf16_t* __restrict__ v, const bf16_t* __restrict__ relT,
    const float* __restrict__ abias, const unsigned char* __restrict__ prm,
    bf16_t* __restrict__ out) {
  __shared__ __align__(16) bf16_t vt[HDIM][SEQ];          // 64 KB: V^T for (b,h)
  __shared__ __align__(16) bf16_t qm[8][NREL][16][HDIM];  // 64 KB: per-wave q@M_r
  __shared__ __align__(16) bf16_t pt[8][16][32];          // 8 KB: per-wave P tile
#if defined(HAVE_TDM)
  __shared__ __align__(16) bf16_t kt[2][32][HDIM];        // 8 KB: K double buffer
#endif

  const int b = blockIdx.z, h = blockIdx.y;
  const int wid = threadIdx.x >> 5, lane = threadIdx.x & 31;
  const int ib = blockIdx.x * 128 + wid * 16;   // first query row (seq idx)
  const int cn = lane & 15, hi = lane >> 4;

  const bf16_t* kbase = k + ((size_t)(b * SEQ)) * DM + h * HDIM;
#if defined(HAVE_TDM)
  if (threadIdx.x < 32)   // wave 0 kicks off first K tile DMA
    tdm_load_2d((uint32_t)(uintptr_t)&kt[0][0][0], kbase, HDIM, 32, (uint32_t)DM,
                (uint32_t)SEQ);
#endif

  // ---- stage V^T: vt[d][j] = v[b, j, h, d]  (u64 global loads, 4 bf16) ----
  for (int idx = threadIdx.x; idx < (HDIM / 4) * SEQ; idx += 256) {
    const int j = idx >> 4, d4 = (idx & 15) * 4;
    V4B val;
    val.u = *(const uint64_t*)&v[((size_t)(b * SEQ + j)) * DM + h * HDIM + d4];
#pragma unroll
    for (int t = 0; t < 4; ++t) vt[d4 + t][j] = val.b[t];
  }

  // ---- q fragments for this wave's 16 rows (K = 64 -> two 16x32 frags) ----
  const bf16_t* qbase = q + ((size_t)(b * SEQ + ib)) * DM + h * HDIM;
  const v16bf qa0 = frag_a(qbase, DM);
  const v16bf qa1 = frag_a(qbase + 32, DM);

  // ---- qm_r = q_tile(16x64) @ M_r(64x64), M_r^T pre-stored as [e][d] ----
#pragma unroll
  for (int r = 0; r < NREL; ++r) {
    const bf16_t* mr = relT + ((size_t)(r * NH + h)) * HDIM * HDIM;
#pragma unroll
    for (int nt = 0; nt < 4; ++nt) {
      F8 c = {};
      v16bf b0 = frag_b(mr + nt * 16 * HDIM, HDIM);
      v16bf b1 = frag_b(mr + nt * 16 * HDIM + 32, HDIM);
      c.v = wmma_bf16(qa0, b0, c.v);
      c.v = wmma_bf16(qa1, b1, c.v);
#pragma unroll
      for (int vv = 0; vv < 8; ++vv)
        qm[wid][r][vv + hi * 8][nt * 16 + cn] = (bf16_t)c.f[vv];
    }
  }
  __syncthreads();   // vt ready for all waves

  float m_run[8], l_run[8];
  F8 oacc[4] = {};
#pragma unroll
  for (int vv = 0; vv < 8; ++vv) { m_run[vv] = -1e30f; l_run[vv] = 0.f; }

  for (int j0 = 0; j0 < SEQ; j0 += 32) {
#if defined(HAVE_TDM)
    const int kbuf = (j0 >> 5) & 1;
    if (threadIdx.x < 32) wait_tensorcnt0();   // current K tile resident
    __syncthreads();
    if (threadIdx.x < 32 && j0 + 32 < SEQ)     // prefetch next K tile
      tdm_load_2d((uint32_t)(uintptr_t)&kt[kbuf ^ 1][0][0],
                  kbase + (size_t)(j0 + 32) * DM, HDIM, 32, (uint32_t)DM,
                  (uint32_t)SEQ);
#endif
    F8 sc[2];
#pragma unroll
    for (int sub = 0; sub < 2; ++sub) {
#if defined(HAVE_TDM)
      const v16bf kb0 = frag_b(&kt[kbuf][sub * 16][0], HDIM);   // head-dim 0..31
      const v16bf kb1 = frag_b(&kt[kbuf][sub * 16][32], HDIM);  // head-dim 32..63
#else
      const bf16_t* kbp = kbase + (size_t)(j0 + sub * 16) * DM;
      const v16bf kb0 = frag_b(kbp, DM);
      const v16bf kb1 = frag_b(kbp + 32, DM);
#endif
      F8 qk = {};
      qk.v = wmma_bf16(qa0, kb0, qk.v);
      qk.v = wmma_bf16(qa1, kb1, qk.v);
      F8 rr[NREL];
#pragma unroll
      for (int r = 0; r < NREL; ++r) {
        v16bf ma0 = frag_a(&qm[wid][r][0][0], HDIM);
        v16bf ma1 = frag_a(&qm[wid][r][0][32], HDIM);
        rr[r].v = v8f{};
        rr[r].v = wmma_bf16(ma0, kb0, rr[r].v);
        rr[r].v = wmma_bf16(ma1, kb1, rr[r].v);
      }
      const int jj = j0 + sub * 16 + cn;
#pragma unroll
      for (int vv = 0; vv < 8; ++vv) {
        const int irow = ib + vv + hi * 8;
        // 4 relation flags are contiguous bytes -> one aligned dword load
        const uint32_t mw =
            *(const uint32_t*)(prm + (((size_t)(b * SEQ + irow)) * SEQ + jj) * NREL);
        float s = qk.f[vv] + (float)(mw & 0xFFu) * rr[0].f[vv] +
                  (float)((mw >> 8) & 0xFFu) * rr[1].f[vv] +
                  (float)((mw >> 16) & 0xFFu) * rr[2].f[vv] +
                  (float)(mw >> 24) * rr[3].f[vv];
        s = s * ATT_SCALE + abias[(((size_t)(b * NH + h)) * SEQ + irow) * SEQ + jj];
        sc[sub].f[vv] = s;
      }
    }

    // ---- online softmax over the 16x32 score block ----
#pragma unroll
    for (int vv = 0; vv < 8; ++vv) {
      float t = fmaxf(sc[0].f[vv], sc[1].f[vv]);
#pragma unroll
      for (int o = 8; o >= 1; o >>= 1) t = fmaxf(t, __shfl_xor(t, o, 32));
      const float nm = fmaxf(m_run[vv], t);
      const float corr = __expf(m_run[vv] - nm);
      m_run[vv] = nm;
      const float p0 = __expf(sc[0].f[vv] - nm);
      const float p1 = __expf(sc[1].f[vv] - nm);
      float ps = p0 + p1;
#pragma unroll
      for (int o = 8; o >= 1; o >>= 1) ps += __shfl_xor(ps, o, 32);
      l_run[vv] = l_run[vv] * corr + ps;
#pragma unroll
      for (int nt = 0; nt < 4; ++nt) oacc[nt].f[vv] *= corr;
      pt[wid][vv + hi * 8][cn] = (bf16_t)p0;
      pt[wid][vv + hi * 8][16 + cn] = (bf16_t)p1;
    }

    // ---- O += P(16x32) @ V(32x64) using staged V^T as B operand ----
    const v16bf pa = frag_a(&pt[wid][0][0], 32);
#pragma unroll
    for (int nt = 0; nt < 4; ++nt) {
      v16bf vbf = frag_b(&vt[nt * 16][j0], SEQ);
      oacc[nt].v = wmma_bf16(pa, vbf, oacc[nt].v);
    }
#if defined(HAVE_TDM)
    __syncthreads();   // all waves done with kt[kbuf] before it is re-filled
#endif
  }

  // ---- normalize and write bf16 head output [b, i, h*64 + d] ----
#pragma unroll
  for (int nt = 0; nt < 4; ++nt)
#pragma unroll
    for (int vv = 0; vv < 8; ++vv) {
      const float val = oacc[nt].f[vv] / l_run[vv];
      out[((size_t)(b * SEQ + ib + vv + hi * 8)) * DM + h * HDIM + nt * 16 + cn] =
          (bf16_t)val;
    }
}

// ---------------------------------------------------------------------------
extern "C" void kernel_launch(void* const* d_in, const int* in_sizes, int n_in,
                              void* d_out, int out_size, void* d_ws, size_t ws_size,
                              hipStream_t stream) {
  const float* x     = (const float*)d_in[0];
  const float* abias = (const float*)d_in[1];
  const unsigned char* prm   = (const unsigned char*)d_in[2];
  const unsigned char* vmask = (const unsigned char*)d_in[3];
  const float* ln1w = (const float*)d_in[4];
  const float* ln1b = (const float*)d_in[5];
  const float* wq = (const float*)d_in[6];  const float* bq = (const float*)d_in[7];
  const float* wk = (const float*)d_in[8];  const float* bk = (const float*)d_in[9];
  const float* wv = (const float*)d_in[10]; const float* bv = (const float*)d_in[11];
  const float* wo = (const float*)d_in[12]; const float* bo = (const float*)d_in[13];
  const float* rel  = (const float*)d_in[14];
  const float* ln2w = (const float*)d_in[15];
  const float* ln2b = (const float*)d_in[16];
  const float* w1 = (const float*)d_in[17]; const float* b1 = (const float*)d_in[18];
  const float* w2 = (const float*)d_in[19]; const float* b2 = (const float*)d_in[20];

  char* ws = (char*)d_ws;
  const size_t MB = 1u << 20;
  bf16_t* xn   = (bf16_t*)(ws + 0 * MB);    // [4096,512]  bf16 LN1(x)
  bf16_t* qb   = (bf16_t*)(ws + 4 * MB);
  bf16_t* kb   = (bf16_t*)(ws + 8 * MB);
  bf16_t* vb   = (bf16_t*)(ws + 12 * MB);
  bf16_t* ao   = (bf16_t*)(ws + 16 * MB);   // attention head-merged output
  bf16_t* xn2  = (bf16_t*)(ws + 20 * MB);
  bf16_t* wqb  = (bf16_t*)(ws + 24 * MB);
  bf16_t* wkb  = (bf16_t*)(ws + 24 * MB + 512 * 1024);
  bf16_t* wvb  = (bf16_t*)(ws + 25 * MB);
  bf16_t* wob  = (bf16_t*)(ws + 25 * MB + 512 * 1024);
  bf16_t* w1b  = (bf16_t*)(ws + 26 * MB);
  bf16_t* w2b  = (bf16_t*)(ws + 28 * MB);
  bf16_t* relb = (bf16_t*)(ws + 30 * MB);
  float*  xmid = (float*)(ws + 31 * MB);    // [4096,512] f32 post-attention x
  bf16_t* hmid = (bf16_t*)(ws + 39 * MB);   // [4096,2048] bf16 GELU mid

  const dim3 blk(256);

  // weight conversion (tiny; hits L2 once)
  cvt_bf16<<<dim3(DM * DM / 256), blk, 0, stream>>>(wq, wqb, DM * DM);
  cvt_bf16<<<dim3(DM * DM / 256), blk, 0, stream>>>(wk, wkb, DM * DM);
  cvt_bf16<<<dim3(DM * DM / 256), blk, 0, stream>>>(wv, wvb, DM * DM);
  cvt_bf16<<<dim3(DM * DM / 256), blk, 0, stream>>>(wo, wob, DM * DM);
  cvt_bf16<<<dim3(FFD * DM / 256), blk, 0, stream>>>(w1, w1b, FFD * DM);
  cvt_bf16<<<dim3(FFD * DM / 256), blk, 0, stream>>>(w2, w2b, FFD * DM);
  cvt_rel<<<dim3(NREL * NH * HDIM * HDIM / 256), blk, 0, stream>>>(rel, relb);

  // LN1 -> xn (bf16)
  ln_bf16<<<dim3(MROWS / 8), blk, 0, stream>>>(x, ln1w, ln1b, xn);

  // QKV projections
  const dim3 gD(DM / 128, MROWS / 128);
  gemm_bf16_wmma<0><<<gD, blk, 0, stream>>>(xn, wqb, bq, qb, nullptr, nullptr, DM, DM);
  gemm_bf16_wmma<0><<<gD, blk, 0, stream>>>(xn, wkb, bk, kb, nullptr, nullptr, DM, DM);
  gemm_bf16_wmma<0><<<gD, blk, 0, stream>>>(xn, wvb, bv, vb, nullptr, nullptr, DM, DM);

  // fused flash attention with relation bilinear bias
  attn_fused<<<dim3(SEQ / 128, NH, BATCH), blk, 0, stream>>>(qb, kb, vb, relb,
                                                             abias, prm, ao);

  // out-proj + residual + valid mask -> xmid (f32)
  gemm_bf16_wmma<1><<<gD, blk, 0, stream>>>(ao, wob, bo, xmid, x, vmask, DM, DM);

  // LN2 -> xn2 (bf16)
  ln_bf16<<<dim3(MROWS / 8), blk, 0, stream>>>(xmid, ln2w, ln2b, xn2);

  // FFN up + exact GELU -> hmid (bf16)
  gemm_bf16_wmma<2><<<dim3(FFD / 128, MROWS / 128), blk, 0, stream>>>(
      xn2, w1b, b1, hmid, nullptr, nullptr, FFD, DM);

  // FFN down + residual + valid mask -> d_out (f32)
  gemm_bf16_wmma<3><<<gD, blk, 0, stream>>>(hmid, w2b, b2, (float*)d_out, xmid,
                                            vmask, DM, FFD);
}
